// CrAKNAttention_87849261072553
// MI455X (gfx1250) — compile-verified, matrix operand order checked by reading, not run on hardware
//
#include <hip/hip_runtime.h>
#include <hip/hip_bf16.h>
#include <math.h>

typedef __attribute__((ext_vector_type(16))) __bf16 v16bf;
typedef __attribute__((ext_vector_type(2)))  __bf16 v2bf;
typedef __attribute__((ext_vector_type(8)))  float  v8f;
typedef __attribute__((ext_vector_type(4)))  float  v4f;

// ---- f32 -> bf16 conversion ------------------------------------------------
#if __has_builtin(__builtin_amdgcn_cvt_pk_bf16_f32)
__device__ __forceinline__ unsigned pack_bf16(float a, float b){
  v2bf p = __builtin_amdgcn_cvt_pk_bf16_f32(a, b);
  return __builtin_bit_cast(unsigned, p);
}
#else
__device__ __forceinline__ unsigned short f32_to_bf16_1(float x){
  unsigned u = __builtin_bit_cast(unsigned, x);
  unsigned r = u + 0x7FFFu + ((u >> 16) & 1u);
  return (unsigned short)(r >> 16);
}
__device__ __forceinline__ unsigned pack_bf16(float a, float b){
  return (unsigned)f32_to_bf16_1(a) | ((unsigned)f32_to_bf16_1(b) << 16);
}
#endif
__device__ __forceinline__ unsigned short f32_to_bf16(float x){
  return (unsigned short)(pack_bf16(x, 0.f) & 0xFFFFu);
}

// Read one lane's 16-bf16 slice of a 16x32 WMMA A/B operand from an LDS row.
// Layout (ISA 7.12.2, 16-bit A 16x32): lanes 0-15 hold K={0..7,16..23},
// lanes 16-31 hold K={8..15,24..31} -> two 16-byte chunks per lane.
__device__ __forceinline__ v16bf lds_read_op(const unsigned short* rowptr, int half){
  const int c0 = half * 8;
  union { v16bf v; uint4 q[2]; } u;
  u.q[0] = *(const uint4*)(rowptr + c0);
  u.q[1] = *(const uint4*)(rowptr + c0 + 16);
  return u.v;
}

// ---------------------------------------------------------------------------
// Generic tiled WMMA GEMM:  C[M,N] = A[M,K] @ B[N,K]^T + bias[N]  (+ mish)
// Block: 256 threads (8 waves), tile 64x64, K-step 32. Requires M%64==N%64==0.
// SQ==1: streaming variant (NT loads of A, NT stores of C) that also
//        accumulates per-(row, 40-col group) sum-of-squares into sqacc via
//        8-lane shfl reduction + one atomic per 8-col chunk (40 == 8 mod 16,
//        so chunk never straddles a group boundary).
// ---------------------------------------------------------------------------
#define GSTRIDE 40   // padded LDS row stride in bf16 (conflict-free b128 reads)

template<int KD, int ACT, int SQ>
__global__ __launch_bounds__(256)
void wmma_gemm_bias_act(const float* __restrict__ A, const float* __restrict__ B,
                        const float* __restrict__ bias, float* __restrict__ C,
                        float* __restrict__ sqacc, int Ndim)
{
  __shared__ unsigned short As[64 * GSTRIDE];
  __shared__ unsigned short Bs[64 * GSTRIDE];

  const int tid  = threadIdx.x;
  const int lane = tid & 31;
  const int wave = tid >> 5;
  const int half = lane >> 4;
  const int l16  = lane & 15;
  const int rg   = wave >> 1;   // row group 0..3 (16 rows each)
  const int cg   = wave & 1;    // col group 0..1 (32 cols each)

  const int row0 = blockIdx.x * 64;
  const int col0 = blockIdx.y * 64;

  v8f acc0 = {}; v8f acc1 = {};

  const int nk = (KD + 31) / 32;
  for (int kt = 0; kt < nk; ++kt){
    const int k0 = kt * 32;
    __syncthreads();
    if (k0 + 32 <= KD){
      // fast path: unguarded float4 loads (global_load_b128)
      for (int i = 0; i < 2; ++i){
        int p = tid + 256 * i;        // 0..511
        int r = p >> 3;               // row 0..63
        int c = (p & 7) * 4;          // col 0,4,..,28
        const v4f* ap = (const v4f*)&A[(size_t)(row0 + r) * KD + k0 + c];
        const v4f av = SQ ? __builtin_nontemporal_load(ap) : *ap;
        *(unsigned*)&As[r * GSTRIDE + c]     = pack_bf16(av[0], av[1]);
        *(unsigned*)&As[r * GSTRIDE + c + 2] = pack_bf16(av[2], av[3]);
        const v4f bv = *(const v4f*)&B[(size_t)(col0 + r) * KD + k0 + c];
        *(unsigned*)&Bs[r * GSTRIDE + c]     = pack_bf16(bv[0], bv[1]);
        *(unsigned*)&Bs[r * GSTRIDE + c + 2] = pack_bf16(bv[2], bv[3]);
      }
    } else {
      // tail: clamped indices + cndmask zeroing, no exec-mask branches
      for (int i = 0; i < 4; ++i){
        int p  = tid + 256 * i;       // 0..1023
        int r  = p >> 4;              // row 0..63
        int c  = (p & 15) * 2;        // col 0,2,..,30
        int ka = k0 + c;
        int i0 = (ka     < KD) ? ka     : KD - 1;
        int i1 = (ka + 1 < KD) ? ka + 1 : KD - 1;
        float a0 = A[(size_t)(row0 + r) * KD + i0]; if (ka     >= KD) a0 = 0.f;
        float a1 = A[(size_t)(row0 + r) * KD + i1]; if (ka + 1 >= KD) a1 = 0.f;
        *(unsigned*)&As[r * GSTRIDE + c] = pack_bf16(a0, a1);
        float b0 = B[(size_t)(col0 + r) * KD + i0]; if (ka     >= KD) b0 = 0.f;
        float b1 = B[(size_t)(col0 + r) * KD + i1]; if (ka + 1 >= KD) b1 = 0.f;
        *(unsigned*)&Bs[r * GSTRIDE + c] = pack_bf16(b0, b1);
      }
    }
    __syncthreads();

    v16bf a  = lds_read_op(&As[(rg * 16 + l16) * GSTRIDE], half);
    v16bf b0 = lds_read_op(&Bs[(cg * 32 +  0 + l16) * GSTRIDE], half);
    v16bf b1 = lds_read_op(&Bs[(cg * 32 + 16 + l16) * GSTRIDE], half);
    acc0 = __builtin_amdgcn_wmma_f32_16x16x32_bf16(false, a, false, b0, (short)0, acc0, false, false);
    acc1 = __builtin_amdgcn_wmma_f32_16x16x32_bf16(false, a, false, b1, (short)0, acc1, false, false);
  }

  for (int ct = 0; ct < 2; ++ct){
    v8f acc = ct ? acc1 : acc0;
    int ncol = col0 + cg * 32 + ct * 16 + l16;
    float bv = bias ? bias[ncol] : 0.f;
    float sq[8];
    for (int r = 0; r < 8; ++r){
      int mrow = row0 + rg * 16 + half * 8 + r;   // C layout: row = r + 8*half
      float x = acc[r] + bv;
      if (ACT == 1){                               // mish
        float sp = (x > 20.f) ? x : log1pf(__expf(x));
        x = x * tanhf(sp);
      }
      float* cp = &C[(size_t)mrow * Ndim + ncol];
      if (SQ) __builtin_nontemporal_store(x, cp); else *cp = x;
      sq[r] = x * x;
    }
    if (SQ){
      // reduce each 8-lane chunk (8 consecutive cols, same row, same group)
      for (int off = 1; off < 8; off <<= 1)
        for (int r = 0; r < 8; ++r)
          sq[r] += __shfl_xor(sq[r], off, 32);
      if ((l16 & 7) == 0){
        int grp = ncol / 40;                       // chunk wholly inside group
        for (int r = 0; r < 8; ++r){
          int mrow = row0 + rg * 16 + half * 8 + r;
          atomicAdd(&sqacc[(size_t)mrow * 8 + grp], sq[r]);
        }
      }
    }
  }
}

// ---------------------------------------------------------------------------
__global__ __launch_bounds__(256)
void zero4(v4f* __restrict__ p, int n4)
{
  int i = blockIdx.x * 256 + threadIdx.x;
  if (i < n4) p[i] = (v4f){0.f, 0.f, 0.f, 0.f};
}

__global__ __launch_bounds__(256)
void sqrt_inplace(float* __restrict__ p, int n)
{
  int i = blockIdx.x * 256 + threadIdx.x;
  if (i < n) p[i] = sqrtf(p[i]);
}

// ---------------------------------------------------------------------------
// Flash attention per head: logits = QK^T/8 + diffs, softmax, O = P V.
// Block = 128 threads (4 waves); each wave owns 16 query rows, block = 64 rows.
// ---------------------------------------------------------------------------
#define AN 768
#define AM 768
#define AH 8
#define AHD 64
#define ASTR 72   // padded LDS row stride in bf16

__global__ __launch_bounds__(128)
void attn_kernel(const float* __restrict__ qh, const float* __restrict__ kh,
                 const float* __restrict__ vh, const float* __restrict__ diffs,
                 float* __restrict__ vals)
{
  __shared__ unsigned short Ks[64 * ASTR];        // K tile, row-major [m][k]
  __shared__ unsigned short Vt[64 * ASTR];        // V tile transposed [d][m]
  __shared__ unsigned short Ps[4 * 16 * ASTR];    // per-wave P staging [q][m]

  const int tid  = threadIdx.x;
  const int lane = tid & 31;
  const int wave = tid >> 5;   // 0..3
  const int half = lane >> 4;
  const int l16  = lane & 15;
  const int h    = blockIdx.y;
  const int q0   = blockIdx.x * 64 + wave * 16;

  // Q A-operand (16 rows x 64 k) straight from global, f32 -> bf16
  v16bf aq[2];
  {
    const float* qp = qh + (size_t)(q0 + l16) * (AH*AHD) + h * AHD;
    for (int ks = 0; ks < 2; ++ks){
      union { v16bf v; unsigned u[8]; } u;
      int base = ks * 32 + half * 8;
      for (int j = 0; j < 4; ++j){
        u.u[j]     = pack_bf16(qp[base + 2*j],      qp[base + 2*j + 1]);
        u.u[j + 4] = pack_bf16(qp[base + 16 + 2*j], qp[base + 16 + 2*j + 1]);
      }
      aq[ks] = u.v;
    }
  }

  v8f o[4] = {};
  float mrun[8], lrun[8];
  for (int r = 0; r < 8; ++r){ mrun[r] = -INFINITY; lrun[r] = 0.f; }

  unsigned short* Pw = &Ps[wave * 16 * ASTR];

  for (int t = 0; t < AM / 64; ++t){
    const int m0 = t * 64;
    __syncthreads();
    // cooperative stage: K tile 64x64, V tile transposed
    for (int i = 0; i < 16; ++i){
      int p = tid + 128 * i;      // 0..2047 pair index
      int r = p >> 5;             // key row 0..63
      int c = (p & 31) * 2;       // col 0,2,..,62
      const float2 kf = *(const float2*)(kh + (size_t)(m0 + r) * (AH*AHD) + h * AHD + c);
      *(unsigned*)&Ks[r * ASTR + c] = pack_bf16(kf.x, kf.y);
      const float2 vf = *(const float2*)(vh + (size_t)(m0 + r) * (AH*AHD) + h * AHD + c);
      Vt[(c    ) * ASTR + r] = f32_to_bf16(vf.x);
      Vt[(c + 1) * ASTR + r] = f32_to_bf16(vf.y);
    }
    __syncthreads();

    // S = Q K^T  (16 x 64 per wave), K dim = 64 -> 2 chained WMMAs per 16-col tile
    v8f s[4];
    for (int ct = 0; ct < 4; ++ct){
      v8f z = {};
      v16bf b0 = lds_read_op(&Ks[(ct*16 + l16) * ASTR     ], half);
      v16bf b1 = lds_read_op(&Ks[(ct*16 + l16) * ASTR + 32], half);
      z = __builtin_amdgcn_wmma_f32_16x16x32_bf16(false, aq[0], false, b0, (short)0, z, false, false);
      z = __builtin_amdgcn_wmma_f32_16x16x32_bf16(false, aq[1], false, b1, (short)0, z, false, false);
      s[ct] = z;
    }

    // scale, add pairwise-bias norms, row max (rows live on 16-lane halves)
    float tmax[8];
    for (int r = 0; r < 8; ++r) tmax[r] = -INFINITY;
    for (int ct = 0; ct < 4; ++ct){
      int m = m0 + ct * 16 + l16;
      for (int r = 0; r < 8; ++r){
        int n = q0 + half * 8 + r;
        float x = s[ct][r] * 0.125f + diffs[((size_t)h * AN + n) * AM + m];
        s[ct][r] = x;
        tmax[r] = fmaxf(tmax[r], x);
      }
    }
    for (int off = 8; off >= 1; off >>= 1)
      for (int r = 0; r < 8; ++r)
        tmax[r] = fmaxf(tmax[r], __shfl_xor(tmax[r], off, 32));

    // online softmax update
    float tsum[8];
    for (int r = 0; r < 8; ++r){
      float mnew  = fmaxf(mrun[r], tmax[r]);
      float scale = (mrun[r] == -INFINITY) ? 0.f : __expf(mrun[r] - mnew);
      mrun[r] = mnew;
      lrun[r] *= scale;
      for (int dt = 0; dt < 4; ++dt) o[dt][r] *= scale;
      float srow = 0.f;
      for (int ct = 0; ct < 4; ++ct){
        float pe = __expf(s[ct][r] - mnew);
        s[ct][r] = pe;
        srow += pe;
      }
      tsum[r] = srow;
    }
    for (int off = 8; off >= 1; off >>= 1)
      for (int r = 0; r < 8; ++r)
        tsum[r] += __shfl_xor(tsum[r], off, 32);
    for (int r = 0; r < 8; ++r) lrun[r] += tsum[r];

    // stage P (C layout) -> LDS -> re-read in A-operand layout
    for (int ct = 0; ct < 4; ++ct)
      for (int r = 0; r < 8; ++r)
        Pw[(half*8 + r) * ASTR + ct*16 + l16] = f32_to_bf16(s[ct][r]);
    __syncthreads();

    v16bf ap0 = lds_read_op(&Pw[l16 * ASTR     ], half);
    v16bf ap1 = lds_read_op(&Pw[l16 * ASTR + 32], half);
    for (int dt = 0; dt < 4; ++dt){
      v16bf bv0 = lds_read_op(&Vt[(dt*16 + l16) * ASTR     ], half);
      v16bf bv1 = lds_read_op(&Vt[(dt*16 + l16) * ASTR + 32], half);
      o[dt] = __builtin_amdgcn_wmma_f32_16x16x32_bf16(false, ap0, false, bv0, (short)0, o[dt], false, false);
      o[dt] = __builtin_amdgcn_wmma_f32_16x16x32_bf16(false, ap1, false, bv1, (short)0, o[dt], false, false);
    }
  }

  // normalize + store vals[n, h*64 + d]
  for (int dt = 0; dt < 4; ++dt){
    int d = dt * 16 + l16;
    for (int r = 0; r < 8; ++r){
      int n = q0 + half * 8 + r;
      vals[(size_t)n * (AH*AHD) + h * AHD + d] = o[dt][r] / lrun[r];
    }
  }
}

// ---------------------------------------------------------------------------
extern "C" void kernel_launch(void* const* d_in, const int* in_sizes, int n_in,
                              void* d_out, int out_size, void* d_ws, size_t ws_size,
                              hipStream_t stream)
{
  const float* q    = (const float*)d_in[0];
  const float* k    = (const float*)d_in[1];
  const float* v    = (const float*)d_in[2];
  const float* bias = (const float*)d_in[3];
  const float* q_w  = (const float*)d_in[4];
  const float* q_b  = (const float*)d_in[5];
  const float* k_w  = (const float*)d_in[6];
  const float* k_b  = (const float*)d_in[7];
  const float* v_w  = (const float*)d_in[8];
  const float* v_b  = (const float*)d_in[9];
  const float* bp_w = (const float*)d_in[10];
  const float* bp_b = (const float*)d_in[11];
  const float* o1_w = (const float*)d_in[12];
  const float* o1_b = (const float*)d_in[13];
  const float* o2_w = (const float*)d_in[14];
  const float* o2_b = (const float*)d_in[15];

  float* out   = (float*)d_out;                 // [768, 512]
  float* biasp = out + (size_t)768 * 512;       // [589824, 320]

  float* ws    = (float*)d_ws;
  float* qh    = ws;                            // [768, 512]
  float* kh    = qh   + 768 * 512;
  float* vh    = kh   + 768 * 512;
  float* vals  = vh   + 768 * 512;
  float* t1    = vals + 768 * 512;
  float* diffs = t1   + 768 * 512;              // [8*768*768] sq-acc -> sqrt in place

  const int NG = 8 * 768 * 768;                 // 4718592 norm groups

  // 0) zero the squared-norm accumulator (re-done every call: deterministic)
  zero4<<<(NG/4 + 255)/256, 256, 0, stream>>>((v4f*)diffs, NG/4);
  // 1) bias_p = bias @ bp_w^T + bp_b  ([589824,40] x [320,40]^T), fused sumsq
  wmma_gemm_bias_act<40, 0, 1><<<dim3(589824/64, 320/64), 256, 0, stream>>>(
      bias, bp_w, bp_b, biasp, diffs, 320);
  // 2) diffs = sqrt(sumsq), in place
  sqrt_inplace<<<(NG + 255)/256, 256, 0, stream>>>(diffs, NG);
  // 3) q/k/v projections
  wmma_gemm_bias_act<512, 0, 0><<<dim3(12, 8), 256, 0, stream>>>(q, q_w, q_b, qh, nullptr, 512);
  wmma_gemm_bias_act<512, 0, 0><<<dim3(12, 8), 256, 0, stream>>>(k, k_w, k_b, kh, nullptr, 512);
  wmma_gemm_bias_act<512, 0, 0><<<dim3(12, 8), 256, 0, stream>>>(v, v_w, v_b, vh, nullptr, 512);
  // 4) attention
  attn_kernel<<<dim3(12, 8), 128, 0, stream>>>(qh, kh, vh, diffs, vals);
  // 5) o1 + mish, 6) o2
  wmma_gemm_bias_act<512, 1, 0><<<dim3(12, 8), 256, 0, stream>>>(vals, o1_w, o1_b, t1, nullptr, 512);
  wmma_gemm_bias_act<512, 0, 0><<<dim3(12, 8), 256, 0, stream>>>(t1, o2_w, o2_b, out, nullptr, 512);
}
